// AttentionMechanism_69449621176565
// MI455X (gfx1250) — matrix operand order repaired and checked by reading, not compiled, not run on hardware
//
#include <hip/hip_runtime.h>
#include <hip/hip_bf16.h>
#include <math.h>

// Problem constants (match reference)
#define BN   4096
#define DN   1024
#define KN   30
#define UN   256
#define CN   128
#define NPAD 96          // 3*K = 90 padded to 6 WMMA N-tiles of 16
#define EPSV 1e-8f
#define KTILES (DN / 32) // 32 k-tiles of depth 32

typedef __attribute__((ext_vector_type(16))) _Float16 v16h;
typedef __attribute__((ext_vector_type(8)))  float    v8f;

// Numerically stable softplus on the TRANS units: v_exp_f32 + v_log_f32,
// branch-free (vs log1pf's ~40-op frexp/ldexp libm expansion).
__device__ __forceinline__ float fast_softplus(float x) {
    return fmaxf(x, 0.0f) + __logf(1.0f + __expf(-fabsf(x)));
}

// -----------------------------------------------------------------------------
// Kernel 0: pre-swizzle W (f32 [D, 90]) into f16 B-fragment layout:
//   Wswz[((kt*6 + nt)*32 + lane)*16 + h]
// where the fragment element (lane, h) maps to W row k = kt*32 + (h/8)*16 + (lane/16)*8 + (h%8),
// column n = nt*16 + lane%16 (zero-padded for n >= 90).
// Each GEMM lane then loads its whole fragment as one aligned 32-byte vector.
// -----------------------------------------------------------------------------
__global__ __launch_bounds__(256) void swizzle_W_kernel(
    const float* __restrict__ W,        // [D, 3K]
    _Float16*    __restrict__ Wswz)     // [KTILES*6*32*16]
{
    const int idx = blockIdx.x * 256 + threadIdx.x;   // 0 .. 98303
    if (idx >= KTILES * 6 * 32 * 16) return;
    const int h    = idx & 15;
    const int lane = (idx >> 4) & 31;
    const int nt   = (idx >> 9) % 6;
    const int kt   = idx / (6 * 32 * 16);

    const int k = kt * 32 + ((h >> 3) << 4) + ((lane >> 4) << 3) + (h & 7);   // B-frag: K = (h/8)*16 + (lane/16)*8 + (h%8)
    const int n = nt * 16 + (lane & 15);              // column = lane % 16
    const float wv = (n < 3 * KN) ? W[(size_t)k * (3 * KN) + n] : 0.0f;
    Wswz[idx] = (_Float16)wv;
}

// -----------------------------------------------------------------------------
// Kernel 1: params = softplus(X @ W + b); kappa = prev_kappa + kappa_inc
// One wave per 16-row tile of X against all 6 N-tiles (N = 96 >= 90).
// A: f32 -> f16 on the fly per ISA 16-bit A 16x32 lane layout (b128 loads +
//    v_cvt_pk_f16_f32). B: one 32-byte vector load per fragment from the
//    pre-swizzled Wswz. Accumulate f32 with v_wmma_f32_16x16x32_f16.
// -----------------------------------------------------------------------------
__global__ __launch_bounds__(32) void gemm_params_kernel(
    const float*    __restrict__ inputs,      // [B, D]
    const _Float16* __restrict__ Wswz,        // swizzled fragments
    const float*    __restrict__ bias,        // [3K]
    const float*    __restrict__ prev_kappa,  // [B, K]
    float* __restrict__ alphaW,               // ws [B, K]
    float* __restrict__ betaW,                // ws [B, K]
    float* __restrict__ kappaW,               // ws [B, K]
    float* __restrict__ out_kappa)            // d_out + B*C, [B, K]
{
    const int lane    = threadIdx.x & 31;
    const int rowBase = blockIdx.x * 16;
    const int lmod    = lane & 15;     // row (A) / column (B/C/D) within tile
    const int lhi     = lane >> 4;     // lane-group select

    v8f acc[6] = {};                   // 6 N-tiles x (16x16 f32 accumulator)

    const float*    arow  = inputs + (size_t)(rowBase + lmod) * DN;
    const _Float16* bbase = Wswz + (size_t)lane * 16;

    for (int kt = 0; kt < KTILES; ++kt) {
        const int k0 = kt * 32;
        // A fragment: lane holds row (rowBase+lmod); half h -> K per ISA table:
        // K = (h/8)*16 + lhi*8 + (h%8)
        v16h afrag;
        #pragma unroll
        for (int h = 0; h < 16; ++h) {
            const int kk = ((h >> 3) << 4) + (lhi << 3) + (h & 7);
            afrag[h] = (_Float16)arow[k0 + kk];
        }
        #pragma unroll
        for (int nt = 0; nt < 6; ++nt) {
            // B fragment: single aligned 32-byte vector load per lane
            const v16h bfrag =
                *(const v16h*)(bbase + ((size_t)(kt * 6 + nt) * 32) * 16);
            acc[nt] = __builtin_amdgcn_wmma_f32_16x16x32_f16(
                /*neg_a=*/false, afrag, /*neg_b=*/false, bfrag,
                /*c_mod=*/(short)0, acc[nt], /*reuse_a=*/false, /*reuse_b=*/false);
        }
    }

    // Epilogue: C/D layout -> VGPR v: row M = v + lhi*8, col N = lmod.
    #pragma unroll
    for (int nt = 0; nt < 6; ++nt) {
        const int j = nt * 16 + lmod;          // column in [0, 96)
        if (j >= 3 * KN) continue;             // padding columns
        const int   pidx = j / KN;             // 0=alpha 1=beta 2=kappa_inc
        const int   cidx = j % KN;
        const float bj   = bias[j];
        #pragma unroll
        for (int v = 0; v < 8; ++v) {
            const int   row = rowBase + (lhi << 3) + v;
            const float x   = acc[nt][v] + bj;
            const float p   = fmaxf(fast_softplus(x), EPSV);
            const size_t idx = (size_t)row * KN + cidx;
            if (pidx == 0)      alphaW[idx] = p;
            else if (pidx == 1) betaW[idx]  = p;
            else {
                const float kap = prev_kappa[idx] + p;
                kappaW[idx]    = kap;
                out_kappa[idx] = kap;
            }
        }
    }
}

// -----------------------------------------------------------------------------
// Kernel 2: one block per batch row.
//   phase 1: thread u computes phi[u] = sum_k alpha*exp(-beta*(kappa-u)^2),
//            masked by sequence length; LDS tree-reduce for the normalizer.
//   phase 2: 2 half-groups x 128 columns stream char_seq row (coalesced 512B
//            per u-step), combine the two U-halves in LDS, write w[b, :].
// This kernel is the bandwidth roof: 512 MB of char_seq @ 23.3 TB/s ~ 22 us.
// -----------------------------------------------------------------------------
__global__ __launch_bounds__(256) void phi_einsum_kernel(
    const float* __restrict__ alphaW,
    const float* __restrict__ betaW,
    const float* __restrict__ kappaW,
    const int*   __restrict__ seqlen,      // [B]
    const float* __restrict__ cs,          // [B, U, C]
    float* __restrict__ out_w)             // [B, C]
{
    __shared__ float s_alpha[KN];
    __shared__ float s_beta[KN];
    __shared__ float s_kappa[KN];
    __shared__ float s_phi[UN];
    __shared__ float s_red[UN];
    __shared__ float s_part[2][CN];

    const int b = blockIdx.x;
    const int t = threadIdx.x;

    if (t < KN) {
        s_alpha[t] = alphaW[(size_t)b * KN + t];
        s_beta[t]  = betaW [(size_t)b * KN + t];
        s_kappa[t] = kappaW[(size_t)b * KN + t];
    }
    __syncthreads();

    const int   L = seqlen[b];
    const float u = (float)t;
    float phi = 0.0f;
    if (t < L) {
        #pragma unroll
        for (int k = 0; k < KN; ++k) {
            const float d = s_kappa[k] - u;
            phi += s_alpha[k] * __expf(-s_beta[k] * d * d);
        }
    }
    s_red[t] = phi;
    __syncthreads();

    // block tree reduction over U=256
    #pragma unroll
    for (int s = 128; s > 0; s >>= 1) {
        if (t < s) s_red[t] += s_red[t + s];
        __syncthreads();
    }
    const float inv = 1.0f / (s_red[0] + EPSV);
    s_phi[t] = phi * inv;
    __syncthreads();

    // phase 2: streaming weighted sum over char positions
    const int c    = t & (CN - 1);
    const int half = t >> 7;                 // which half of U this thread sums
    const float* csrow = cs + (size_t)b * UN * CN;
    const int u0 = half * (UN / 2);
    float acc = 0.0f;
    #pragma unroll 4
    for (int uu = 0; uu < UN / 2; ++uu) {
        acc += s_phi[u0 + uu] * csrow[(size_t)(u0 + uu) * CN + c];
    }
    s_part[half][c] = acc;
    __syncthreads();
    if (t < CN) out_w[(size_t)b * CN + t] = s_part[0][t] + s_part[1][t];
}

// -----------------------------------------------------------------------------
extern "C" void kernel_launch(void* const* d_in, const int* in_sizes, int n_in,
                              void* d_out, int out_size, void* d_ws, size_t ws_size,
                              hipStream_t stream) {
    const float* inputs     = (const float*)d_in[0];  // [B, D]
    const float* prev_kappa = (const float*)d_in[1];  // [B, K]
    const float* char_seq   = (const float*)d_in[2];  // [B, U, C]
    const int*   seq_len    = (const int*)  d_in[3];  // [B]
    const float* W          = (const float*)d_in[4];  // [D, 3K]
    const float* bias       = (const float*)d_in[5];  // [3K]

    float* out   = (float*)d_out;
    float* out_w = out;                   // [B, C]
    float* out_k = out + (size_t)BN * CN; // [B, K]

    // workspace: Wswz (f16) | alpha | beta | kappa (each [B, K] f32)
    _Float16* Wswz   = (_Float16*)d_ws;
    float*    alphaW = (float*)((char*)d_ws + (size_t)KTILES * 6 * 32 * 16 * sizeof(_Float16));
    float*    betaW  = alphaW + (size_t)BN * KN;
    float*    kappaW = betaW  + (size_t)BN * KN;

    // Kernel 0: swizzle W into fragment-major f16 layout (98304 elements)
    swizzle_W_kernel<<<(KTILES * 6 * 32 * 16 + 255) / 256, 256, 0, stream>>>(W, Wswz);

    // Kernel 1: one wave per 16-row tile -> 256 blocks spread across WGPs
    gemm_params_kernel<<<BN / 16, 32, 0, stream>>>(
        inputs, Wswz, bias, prev_kappa, alphaW, betaW, kappaW, out_k);

    // Kernel 2: one block per batch row
    phi_einsum_kernel<<<BN, 256, 0, stream>>>(
        alphaW, betaW, kappaW, seq_len, char_seq, out_w);
}
